// SS2D_55894704390739
// MI455X (gfx1250) — matrix operand order repaired and checked by reading, not compiled, not run on hardware
//
#include <hip/hip_runtime.h>
#include <hip/hip_bf16.h>

// ---------------------------------------------------------------------------
// SS2D (2D selective scan) for gfx1250 / MI455X.
// f16 WMMA (v_wmma_f32_16x16x32_f16) for all GEMMs, f32 VALU for the scan,
// GLOBAL_LOAD_ASYNC_TO_LDS_B128 (ASYNCcnt) for bulk global->LDS staging.
// ---------------------------------------------------------------------------

typedef __attribute__((ext_vector_type(16))) _Float16 v16h;
typedef __attribute__((ext_vector_type(8)))  float    v8f;
typedef __attribute__((ext_vector_type(4)))  int      v4i;

#define DEV __device__ __forceinline__

DEV int imin_(int a, int b){ return a < b ? a : b; }
DEV int imax_(int a, int b){ return a > b ? a : b; }
DEV float sigmoid_(float x){ return 1.0f / (1.0f + expf(-x)); }
DEV float softplus_(float x){ return x > 20.0f ? x : log1pf(expf(x)); }

// ---- async global -> LDS copy (CDNA5: tracked by ASYNCcnt) -----------------
#if defined(__gfx1250__) && __has_builtin(__builtin_amdgcn_global_load_async_to_lds_b128)
#define HAS_ASYNC_LDS 1
#else
#define HAS_ASYNC_LDS 0
#endif

#if HAS_ASYNC_LDS
typedef __attribute__((address_space(1))) v4i* gptr_v4i;
typedef __attribute__((address_space(3))) v4i* lptr_v4i;
#endif

DEV void async_ld_b128(const void* g, void* l) {
#if HAS_ASYNC_LDS
  __builtin_amdgcn_global_load_async_to_lds_b128((gptr_v4i)g, (lptr_v4i)l, 0, 0);
#else
  *(float4*)l = *(const float4*)g;
#endif
}

DEV void async_wait() {
#if HAS_ASYNC_LDS
#if __has_builtin(__builtin_amdgcn_s_wait_asynccnt)
  __builtin_amdgcn_s_wait_asynccnt(0);
#else
  asm volatile("s_wait_asynccnt 0" ::: "memory");
#endif
#endif
}

// ---- WMMA fragment helpers (contiguous k order chosen here, A and B match, wave32) ----------
// A (16x32 f16): lane holds row m = lane%16; element i -> k = (lane/16)*16 + i
// within the 32-wide K tile.
DEV v16h load_A(const _Float16* base, int row0, int k0, int ld, int lane) {
  int m = row0 + (lane & 15);
  const _Float16* r = base + m * ld + k0 + ((lane >> 4) & 1) * 16;
  v16h a;
#pragma unroll
  for (int i = 0; i < 16; ++i) a[i] = r[i];
  return a;
}

// B (32x16 f16) from a weight matrix stored row-major as W[n][k]
// (so B[k][n] = W[n][k]): lane holds column n = lane%16,
// element j -> k = (lane/16)*16 + j.
DEV v16h load_Bw(const _Float16* wt, int n0, int k0, int ld, int lane) {
  const _Float16* r = wt + (size_t)(n0 + (lane & 15)) * ld + k0 + ((lane >> 4) & 1) * 16;
  v16h b;
#pragma unroll
  for (int j = 0; j < 16; ++j) b[j] = r[j];
  return b;
}

DEV v8f wmma_f16(v16h a, v16h b, v8f c) {
  return __builtin_amdgcn_wmma_f32_16x16x32_f16(false, a, false, b, (short)0, c, false, false);
}

// ---------------------------------------------------------------------------
// Weight prep: fp32 -> f16 copies (x-proj padded 36 -> 48 rows for WMMA N).
// ---------------------------------------------------------------------------
__global__ void k_prep(const float* __restrict__ in_w, const float* __restrict__ xp_w,
                       const float* __restrict__ out_w, const float* __restrict__ w1,
                       _Float16* __restrict__ in_w_h, _Float16* __restrict__ xp_h,
                       _Float16* __restrict__ out_w_h, _Float16* __restrict__ w1_h) {
  const int n_in = 8 * 256 * 64;
  const int n_xp = 8 * 48 * 128;
  const int n_ow = 8 * 64 * 128;
  const int n_w1 = 16 * 512;
  int i = blockIdx.x * blockDim.x + threadIdx.x;
  if (i < n_in) {
    in_w_h[i] = (_Float16)in_w[i];
  } else if (i < n_in + n_xp) {
    int j = i - n_in;
    int sl = j / (48 * 128); int r = (j / 128) % 48; int c = j % 128;
    xp_h[j] = (r < 36) ? (_Float16)xp_w[((size_t)sl * 36 + r) * 128 + c] : (_Float16)0.0f;
  } else if (i < n_in + n_xp + n_ow) {
    int j = i - n_in - n_xp;
    out_w_h[j] = (_Float16)out_w[j];
  } else if (i < n_in + n_xp + n_ow + n_w1) {
    int j = i - n_in - n_xp - n_ow;
    w1_h[j] = (_Float16)w1[j];
  }
}

// (B,C,H,W) -> (B,HW,C)
__global__ void k_tr(const float* __restrict__ x, float* __restrict__ x_hw) {
  int i = blockIdx.x * blockDim.x + threadIdx.x;
  if (i >= 2 * 64 * 4096) return;
  int b = i / (64 * 4096); int c = (i / 4096) % 64; int hw = i % 4096;
  x_hw[((size_t)b * 4096 + hw) * 64 + c] = x[i];
}

// Diagonal position map (H=W=64): returns hw index or 4096 (pad)
DEV int diag_idx(int i, int t, bool anti, bool rev) {
  int cnt = imin_(i, 126 - i) + 1;
  if (t >= cnt) return 4096;
  if (rev) t = cnt - 1 - t;
  int jmin = anti ? imax_(0, 63 - i) : imax_(0, i - 63);
  int j = jmin + t;
  int k = anti ? (i - 63 + j) : (i - j);
  return (j << 6) + k;
}

DEV int dir_hw(int dir, int sIdx, int t, int* b_out) {
  if (dir < 4) {
    int b = sIdx >> 6; int p = sIdx & 63; *b_out = b;
    int h, w;
    if (dir == 0)      { h = p;      w = t; }
    else if (dir == 1) { h = p;      w = 63 - t; }
    else if (dir == 2) { w = p;      h = t; }
    else               { w = p;      h = 63 - t; }
    return (h << 6) + w;
  } else {
    int b = sIdx / 127; int d = sIdx % 127; *b_out = b;
    return diag_idx(d, t, dir >= 6, (dir & 1) != 0);
  }
}

__global__ void k_gather(const float* __restrict__ x_hw, _Float16* __restrict__ seq,
                         int dir, int nseq) {
  int i = blockIdx.x * blockDim.x + threadIdx.x;
  if (i >= nseq * 4096) return;
  int c = i & 63; int t = (i >> 6) & 63; int sIdx = i >> 12;
  int b; int hw = dir_hw(dir, sIdx, t, &b);
  float v = (hw < 4096) ? x_hw[((size_t)b * 4096 + hw) * 64 + c] : 0.0f;
  seq[i] = (_Float16)v;
}

__global__ void k_scatter(const _Float16* __restrict__ seq, float* __restrict__ dirbuf,
                          int dir, int nseq) {
  int i = blockIdx.x * blockDim.x + threadIdx.x;
  if (i >= nseq * 4096) return;
  int c = i & 63; int t = (i >> 6) & 63; int sIdx = i >> 12;
  int b; int hw = dir_hw(dir, sIdx, t, &b);
  if (hw < 4096) dirbuf[((size_t)b * 4096 + hw) * 64 + c] = (float)seq[i];
}

// ---------------------------------------------------------------------------
// One Mamba layer + RMSNorm. One workgroup (4 wave32) per sequence (L=64,C=64).
// Activations between layers are f16 (WMMA A operand precision anyway).
// ---------------------------------------------------------------------------
__global__ __launch_bounds__(128) void k_mamba(
    const _Float16* __restrict__ seq_in, _Float16* __restrict__ seq_out,
    const _Float16* __restrict__ in_w_h,   // 256x64
    const float*    __restrict__ conv_w,   // 128x4
    const float*    __restrict__ conv_b,   // 128
    const _Float16* __restrict__ xp_h,     // 48x128 (rows 36..47 zero)
    const float*    __restrict__ dt_w,     // 128x4
    const float*    __restrict__ dt_b,     // 128
    const float*    __restrict__ A_log,    // 128x16
    const float*    __restrict__ Dsk,      // 128
    const _Float16* __restrict__ out_w_h,  // 64x128
    const float*    __restrict__ norm_w)   // 64
{
  __shared__ __align__(16) _Float16 s_x[64 * 64];   //  8 KB input (async-staged)
  __shared__ _Float16 s_u[64 * 128];    // 16 KB  pre-conv u
  __shared__ _Float16 s_z[64 * 128];    // 16 KB  gate z
  __shared__ float    s_uc[64 * 128];   // 32 KB  post-conv uc (later: scan y)
  __shared__ _Float16 s_uch[64 * 128];  // 16 KB  f16 uc (later: gated y2)
  __shared__ float    s_dbc[64 * 36];   //  9 KB  dt-raw | B | C
  __shared__ float    s_dt[64 * 128];   // 32 KB  dt (later reused as out f32)

  const int tid  = threadIdx.x;
  const int lane = tid & 31;
  const int wv   = tid >> 5;

  // ---- async-stage input tile (8 KB) into LDS -----------------------------
  {
    const _Float16* src = seq_in + (size_t)blockIdx.x * 4096;
#pragma unroll
    for (int i = tid * 8; i < 4096; i += 128 * 8)
      async_ld_b128(src + i, &s_x[i]);
    async_wait();
  }
  __syncthreads();

  // ---- in-proj: xz(64x256) = X(64x64) @ in_w^T, split u|z -----------------
  {
    const int mb = wv;
    for (int nb = 0; nb < 16; ++nb) {
      v8f acc = {};
#pragma unroll
      for (int kb = 0; kb < 2; ++kb)
        acc = wmma_f16(load_A(s_x, mb * 16, kb * 32, 64, lane),
                       load_Bw(in_w_h, nb * 16, kb * 32, 64, lane), acc);
      int n  = nb * 16 + (lane & 15);
      int m0 = mb * 16 + ((lane >> 4) & 1) * 8;
      _Float16* dst = (n < 128) ? s_u : s_z;
      int nn = n & 127;
#pragma unroll
      for (int r = 0; r < 8; ++r) dst[(m0 + r) * 128 + nn] = (_Float16)acc[r];
    }
  }
  __syncthreads();

  // ---- causal depthwise conv(4) + SiLU, thread = channel ------------------
  {
    const int d = tid;
    float w0 = conv_w[d * 4 + 0], w1 = conv_w[d * 4 + 1];
    float w2 = conv_w[d * 4 + 2], w3 = conv_w[d * 4 + 3];
    float bb = conv_b[d];
    float u0 = 0.f, u1 = 0.f, u2 = 0.f;
    for (int t = 0; t < 64; ++t) {
      float u3 = (float)s_u[t * 128 + d];
      float a = bb + u0 * w0 + u1 * w1 + u2 * w2 + u3 * w3;
      float v = a * sigmoid_(a);
      s_uc[t * 128 + d]  = v;
      s_uch[t * 128 + d] = (_Float16)v;
      u0 = u1; u1 = u2; u2 = u3;
    }
  }
  __syncthreads();

  // ---- x-proj: dbc(64x36) = uc(64x128) @ xp^T (padded to 48) --------------
  {
    const int mb = wv;
    for (int nb = 0; nb < 3; ++nb) {
      v8f acc = {};
#pragma unroll
      for (int kb = 0; kb < 4; ++kb)
        acc = wmma_f16(load_A(s_uch, mb * 16, kb * 32, 128, lane),
                       load_Bw(xp_h, nb * 16, kb * 32, 128, lane), acc);
      int n  = nb * 16 + (lane & 15);
      int m0 = mb * 16 + ((lane >> 4) & 1) * 8;
      if (n < 36) {
#pragma unroll
        for (int r = 0; r < 8; ++r) s_dbc[(m0 + r) * 36 + n] = acc[r];
      }
    }
  }
  __syncthreads();

  // ---- dt = softplus(dbc[:, :4] @ dt_w^T + dt_b) --------------------------
  {
    const int d = tid;
    float q0 = dt_w[d * 4 + 0], q1 = dt_w[d * 4 + 1];
    float q2 = dt_w[d * 4 + 2], q3 = dt_w[d * 4 + 3];
    float bb = dt_b[d];
    for (int t = 0; t < 64; ++t) {
      const float* r = s_dbc + t * 36;
      s_dt[t * 128 + d] = softplus_(bb + r[0] * q0 + r[1] * q1 + r[2] * q2 + r[3] * q3);
    }
  }
  __syncthreads();

  // ---- selective scan (16 states in registers, thread = channel) ----------
  {
    const int d = tid;
    float A[16], h[16];
#pragma unroll
    for (int s = 0; s < 16; ++s) { A[s] = -expf(A_log[d * 16 + s]); h[s] = 0.f; }
    const float Dd = Dsk[d];
    for (int t = 0; t < 64; ++t) {
      float dtv = s_dt[t * 128 + d];
      float ucv = s_uc[t * 128 + d];
      float du  = dtv * ucv;
      const float* bc = s_dbc + t * 36;
      float y = 0.f;
#pragma unroll
      for (int s = 0; s < 16; ++s) {
        h[s] = h[s] * expf(dtv * A[s]) + du * bc[4 + s];
        y   += h[s] * bc[20 + s];
      }
      s_uc[t * 128 + d] = y + ucv * Dd;   // overwrite uc with scan output
    }
  }
  __syncthreads();

  // ---- gate: y2 = y * silu(z) ---------------------------------------------
  {
    const int d = tid;
    for (int t = 0; t < 64; ++t) {
      float z = (float)s_z[t * 128 + d];
      s_uch[t * 128 + d] = (_Float16)(s_uc[t * 128 + d] * z * sigmoid_(z));
    }
  }
  __syncthreads();

  // ---- out-proj: out(64x64) = y2(64x128) @ out_w^T ------------------------
  float* s_out = s_dt;  // reuse (dt is dead), 64x64 f32
  {
    const int mb = wv;
    for (int nb = 0; nb < 4; ++nb) {
      v8f acc = {};
#pragma unroll
      for (int kb = 0; kb < 4; ++kb)
        acc = wmma_f16(load_A(s_uch, mb * 16, kb * 32, 128, lane),
                       load_Bw(out_w_h, nb * 16, kb * 32, 128, lane), acc);
      int n  = nb * 16 + (lane & 15);
      int m0 = mb * 16 + ((lane >> 4) & 1) * 8;
#pragma unroll
      for (int r = 0; r < 8; ++r) s_out[(m0 + r) * 64 + n] = acc[r];
    }
  }
  __syncthreads();

  // ---- RMSNorm + writeout (f16 activations between layers) ----------------
  if (tid < 64) {
    const float* r = s_out + tid * 64;
    float ss = 0.f;
    for (int c = 0; c < 64; ++c) ss += r[c] * r[c];
    float inv = rsqrtf(ss * (1.0f / 64.0f) + 1e-5f);
    _Float16* o = seq_out + (size_t)blockIdx.x * 4096 + tid * 64;
    for (int c = 0; c < 64; ++c) o[c] = (_Float16)(r[c] * inv * norm_w[c]);
  }
}

// ---------------------------------------------------------------------------
// Attention fusion: 16 pixels per block; a1 = relu(fused@W1^T+b1) via WMMA.
// Fused tile (32 KB) is async-staged into LDS.
// ---------------------------------------------------------------------------
__global__ __launch_bounds__(64) void k_fuse(
    const float* __restrict__ dirs,      // [8][B][4096][64]
    const float* __restrict__ x_hw,      // [B][4096][64]
    const _Float16* __restrict__ w1_h,   // 16x512
    const float* __restrict__ b1,        // 16
    const float* __restrict__ w2,        // 8x16
    const float* __restrict__ b2,        // 8
    float* __restrict__ out)             // [B][64][4096]
{
  __shared__ __align__(16) float s_ff[16 * 512];  // 32 KB fused f32
  __shared__ _Float16 s_fh[16 * 512];             // 16 KB fused f16 (WMMA A)
  __shared__ float    s_a1[16 * 16];
  __shared__ float    s_w[16 * 8];
  const int tid = threadIdx.x;
  const int lane = tid & 31;
  const int p0 = blockIdx.x * 16;
  const int b   = p0 >> 12;       // block never straddles a batch boundary
  const int hw0 = p0 & 4095;

  // async-stage fused directions: s_ff[p][s*64+c] = dirs[s][b][hw0+p][c]
  for (int e4 = tid; e4 < 16 * 128; e4 += 64) {
    int e = e4 * 4;                       // element index, 4 floats per chunk
    int p = e >> 9; int f = e & 511;
    int s = f >> 6; int c = f & 63;
    async_ld_b128(&dirs[(((size_t)s * 2 + b) * 4096 + hw0 + p) * 64 + c], &s_ff[e]);
  }
  async_wait();
  __syncthreads();

  for (int e = tid; e < 16 * 512; e += 64) s_fh[e] = (_Float16)s_ff[e];
  __syncthreads();

  if (tid < 32) {  // wave 0 does the 16x16 (K=512) GEMM
    v8f acc = {};
    for (int kb = 0; kb < 16; ++kb)
      acc = wmma_f16(load_A(s_fh, 0, kb * 32, 512, lane),
                     load_Bw(w1_h, 0, kb * 32, 512, lane), acc);
    int n = lane & 15; int m0 = (lane >> 4) * 8;
#pragma unroll
    for (int r = 0; r < 8; ++r) s_a1[(m0 + r) * 16 + n] = acc[r];
  }
  __syncthreads();

  if (tid < 16) {
    const int p = tid;
    float a2[8];
    for (int d = 0; d < 8; ++d) {
      float acc = b2[d];
      for (int o = 0; o < 16; ++o)
        acc += fmaxf(0.f, s_a1[p * 16 + o] + b1[o]) * w2[d * 16 + o];
      a2[d] = acc;
    }
    float m = a2[0];
    for (int d = 1; d < 8; ++d) m = fmaxf(m, a2[d]);
    float sum = 0.f;
    for (int d = 0; d < 8; ++d) { a2[d] = expf(a2[d] - m); sum += a2[d]; }
    float inv = 1.f / sum;
    for (int d = 0; d < 8; ++d) s_w[p * 8 + d] = a2[d] * inv;
  }
  __syncthreads();

  for (int e = tid; e < 16 * 64; e += 64) {
    int p = e >> 6; int c = e & 63;
    int gp = p0 + p;
    float acc = x_hw[(size_t)gp * 64 + c];
#pragma unroll
    for (int s = 0; s < 8; ++s) acc += s_w[p * 8 + s] * s_ff[p * 512 + s * 64 + c];
    out[((size_t)b * 64 + c) * 4096 + (hw0 + p)] = acc;
  }
}

// ---------------------------------------------------------------------------
extern "C" void kernel_launch(void* const* d_in, const int* in_sizes, int n_in,
                              void* d_out, int out_size, void* d_ws, size_t ws_size,
                              hipStream_t stream) {
  (void)in_sizes; (void)n_in; (void)out_size; (void)ws_size;
  const float* x       = (const float*)d_in[0];
  const float* in_w    = (const float*)d_in[1];
  const float* conv_w  = (const float*)d_in[2];
  const float* conv_b  = (const float*)d_in[3];
  const float* xproj_w = (const float*)d_in[4];
  const float* dt_w    = (const float*)d_in[5];
  const float* dt_b    = (const float*)d_in[6];
  const float* A_log   = (const float*)d_in[7];
  const float* Dskip   = (const float*)d_in[8];
  const float* out_w   = (const float*)d_in[9];
  const float* norm_w  = (const float*)d_in[10];
  const float* attn_w1 = (const float*)d_in[11];
  const float* attn_b1 = (const float*)d_in[12];
  const float* attn_w2 = (const float*)d_in[13];
  const float* attn_b2 = (const float*)d_in[14];
  float* outp = (float*)d_out;

  char* ws = (char*)d_ws;
  size_t off = 0;
  auto alloc = [&](size_t bytes) -> char* {
    char* p = ws + off; off += (bytes + 255) & ~(size_t)255; return p;
  };
  float*    x_hw    = (float*)   alloc(sizeof(float) * 2 * 4096 * 64);
  float*    dirs    = (float*)   alloc(sizeof(float) * 8 * 2 * 4096 * 64);
  _Float16* seqA    = (_Float16*)alloc(sizeof(_Float16) * 508 * 4096);
  _Float16* seqB    = (_Float16*)alloc(sizeof(_Float16) * 508 * 4096);
  _Float16* in_w_h  = (_Float16*)alloc(sizeof(_Float16) * 8 * 256 * 64);
  _Float16* xp_h    = (_Float16*)alloc(sizeof(_Float16) * 8 * 48 * 128);
  _Float16* out_w_h = (_Float16*)alloc(sizeof(_Float16) * 8 * 64 * 128);
  _Float16* w1_h    = (_Float16*)alloc(sizeof(_Float16) * 16 * 512);

  {
    int total = 8 * 256 * 64 + 8 * 48 * 128 + 8 * 64 * 128 + 16 * 512;
    k_prep<<<(total + 255) / 256, 256, 0, stream>>>(in_w, xproj_w, out_w, attn_w1,
                                                    in_w_h, xp_h, out_w_h, w1_h);
  }
  {
    int total = 2 * 64 * 4096;
    k_tr<<<(total + 255) / 256, 256, 0, stream>>>(x, x_hw);
  }

  // Process direction pairs together (they share a parameter stack).
  for (int pr = 0; pr < 4; ++pr) {
    int d0 = 2 * pr, d1 = 2 * pr + 1;
    int nseq = (d0 < 4) ? 128 : 254;
    int tot  = nseq * 4096;
    k_gather<<<(tot + 255) / 256, 256, 0, stream>>>(x_hw, seqA, d0, nseq);
    k_gather<<<(tot + 255) / 256, 256, 0, stream>>>(x_hw, seqA + (size_t)tot, d1, nseq);
    for (int l = 0; l < 2; ++l) {
      int sl = pr * 2 + l;
      const _Float16* src = (l == 0) ? seqA : seqB;
      _Float16*       dst = (l == 0) ? seqB : seqA;
      k_mamba<<<2 * nseq, 128, 0, stream>>>(
          src, dst,
          in_w_h + (size_t)sl * 256 * 64,
          conv_w + (size_t)sl * 128 * 4,
          conv_b + (size_t)sl * 128,
          xp_h + (size_t)sl * 48 * 128,
          dt_w + (size_t)sl * 128 * 4,
          dt_b + (size_t)sl * 128,
          A_log + (size_t)sl * 128 * 16,
          Dskip + (size_t)sl * 128,
          out_w_h + (size_t)sl * 64 * 128,
          norm_w + (size_t)sl * 64);
    }
    k_scatter<<<(tot + 255) / 256, 256, 0, stream>>>(seqA, dirs + (size_t)d0 * 2 * 4096 * 64, d0, nseq);
    k_scatter<<<(tot + 255) / 256, 256, 0, stream>>>(seqA + (size_t)tot, dirs + (size_t)d1 * 2 * 4096 * 64, d1, nseq);
  }

  k_fuse<<<8192 / 16, 64, 0, stream>>>(dirs, x_hw, w1_h, attn_b1, attn_w2, attn_b2, outp);
}
